// Attention_41755672052175
// MI455X (gfx1250) — matrix operand order repaired and checked by reading, not compiled, not run on hardware
//
#include <hip/hip_runtime.h>

// Problem constants (match reference): B=4, D=512, L=2048, H=8, hd=64
#define B_   4
#define D_   512
#define L_   2048
#define H_   8
#define HD_  64

typedef __attribute__((ext_vector_type(16))) __bf16 v16bf;
typedef __attribute__((ext_vector_type(8)))  __bf16 v8bf;
typedef __attribute__((ext_vector_type(8)))  float  v8f;
typedef __attribute__((ext_vector_type(4)))  float  v4f;
typedef __attribute__((ext_vector_type(4)))  unsigned int v4u;
typedef __attribute__((ext_vector_type(8)))  unsigned int v8u;

__device__ __forceinline__ v16bf cat8(v8bf lo, v8bf hi) {
  v16bf r;
#pragma unroll
  for (int e = 0; e < 8; ++e) { r[e] = lo[e]; r[e + 8] = hi[e]; }
  return r;
}

__device__ __forceinline__ v8f wmma_bf16(v16bf a, v16bf b, v8f c) {
  // D = A(16x32 bf16) * B(32x16 bf16) + C(16x16 f32)
  return __builtin_amdgcn_wmma_f32_16x16x32_bf16(false, a, false, b,
                                                 (short)0, c, false, false);
}

// Low 32 bits of a flat LDS pointer == LDS byte offset (ISA aperture rules).
__device__ __forceinline__ unsigned ldsoff(const void* p) {
  return (unsigned)(size_t)p;
}

// ---------------------------------------------------------------------------
// TDM: DMA a [128 rows x 64 bf16] tile (row stride 64 elems in memory) into
// LDS with hardware padding of 4 DWORDs every 32 DWORDs -> LDS row stride of
// 72 bf16, matching the padded __shared__ tiles below. 2D tensor -> groups
// 0/1 only. Issued by one wave; tracked with TENSORcnt.
// ---------------------------------------------------------------------------
__device__ __forceinline__ void tdm_load_tile_128x64(const __bf16* gsrc,
                                                     unsigned lds_byte_off) {
  unsigned long long ga = (unsigned long long)(size_t)gsrc;
  v4u g0;
  g0[0] = 1u;                                    // count=1, user descriptor
  g0[1] = lds_byte_off;                          // lds_addr
  g0[2] = (unsigned)(ga & 0xffffffffu);          // global_addr[31:0]
  g0[3] = (unsigned)((ga >> 32) & 0x01ffffffu)   // global_addr[56:32]
          | 0x80000000u;                         // type=2 ("image")
  v8u g1;
  g1[0] = (1u << 16)      // data_size = 2 bytes (bf16)
        | (1u << 20)      // pad_enable
        | (4u << 22)      // pad_interval: 32 DWORDs (=128B row)
        | (3u << 25);     // pad_amount:   4 DWORDs (=8 bf16 pad)
  g1[1] = 64u << 16;      // tensor_dim0 = 64   (bits 79:48)
  g1[2] = 128u << 16;     // tensor_dim1 = 128  (bits 111:80)
  g1[3] = 64u << 16;      // tile_dim0  = 64    (bits 127:112)
  g1[4] = 128u;           // tile_dim1  = 128   (bits 143:128)
  g1[5] = 64u;            // tensor_dim0_stride = 64 (bits 207:160)
  g1[6] = 0u;
  g1[7] = 0u;
  asm volatile("tensor_load_to_lds %0, %1" :: "s"(g0), "s"(g1) : "memory");
}

// ---------------------------------------------------------------------------
// Stage 1: qkv[b,o,l] = sum_d w[o,d] * x[b,d,l]   (W[1536x512] x X[512x2048])
// Written transposed per head: qT/kT/vT [bh][l][64] in bf16; q pre-scaled.
// WG: 256 thr = 8 waves, tile 128(M=o) x 64(N=l); wave tile 32x32 (2x2 WMMA).
// ---------------------------------------------------------------------------
__global__ __launch_bounds__(256) void qkv_gemm_kernel(
    const float* __restrict__ x, const float* __restrict__ w,
    __bf16* __restrict__ qT, __bf16* __restrict__ kT, __bf16* __restrict__ vT)
{
  __shared__ __bf16 sW[128][40];   // A tile [o][d], pad 32->40
  __shared__ __bf16 sX[64][40];    // B tile stored [l][d] so k-pairs contiguous

  const int b  = blockIdx.z;
  const int m0 = blockIdx.y * 128;
  const int n0 = blockIdx.x * 64;
  const int tid  = threadIdx.x;
  const int lane = tid & 31, wv = tid >> 5;
  const int g = lane >> 4, ln = lane & 15;
  const int wm = (wv >> 1) * 32;   // 4 waves along M
  const int wn = (wv & 1) * 32;    // 2 waves along N

  v8f acc[2][2];
#pragma unroll
  for (int i = 0; i < 2; ++i)
#pragma unroll
    for (int j = 0; j < 2; ++j)
#pragma unroll
      for (int e = 0; e < 8; ++e) acc[i][j][e] = 0.f;

  for (int k0 = 0; k0 < D_; k0 += 32) {
    // W tile 128x32 f32 -> bf16 (vectorized float4 reads)
#pragma unroll
    for (int it = 0; it < 4; ++it) {
      int i = tid + it * 256;
      int mm = i >> 3, q4 = i & 7;
      v4f v = *(const v4f*)(w + (size_t)(m0 + mm) * D_ + k0 + q4 * 4);
#pragma unroll
      for (int e = 0; e < 4; ++e) sW[mm][q4 * 4 + e] = (__bf16)v[e];
    }
    // X tile 32x64, transposed into sX[l][d]
#pragma unroll
    for (int it = 0; it < 2; ++it) {
      int i = tid + it * 256;
      int kk = i >> 4, c = i & 15;
      v4f v = *(const v4f*)(x + ((size_t)b * D_ + k0 + kk) * L_ + n0 + c * 4);
#pragma unroll
      for (int e = 0; e < 4; ++e) sX[c * 4 + e][kk] = (__bf16)v[e];
    }
    __syncthreads();

#pragma unroll
    for (int fi = 0; fi < 2; ++fi) {
      // A 16x32: lane row = ln, elems 0..7 -> k=g*8+0..7, 8..15 -> 16+g*8+0..7
      const __bf16* pa = &sW[wm + 16 * fi + ln][g * 8];
      v16bf a = cat8(*(const v8bf*)pa, *(const v8bf*)(pa + 16));
#pragma unroll
      for (int fj = 0; fj < 2; ++fj) {
        // B 32x16: lane col = ln, elems 0..7 -> k=g*8+0..7, 8..15 -> 16+g*8+0..7
        const __bf16* pb = &sX[wn + 16 * fj + ln][g * 8];
        v16bf bb = cat8(*(const v8bf*)pb, *(const v8bf*)(pb + 16));
        acc[fi][fj] = wmma_bf16(a, bb, acc[fi][fj]);
      }
    }
    __syncthreads();
  }

  // Epilogue. Each 128-row tile lies wholly in one q/k/v segment
  // (512 % 128 == 0) -> segment select is BLOCK-UNIFORM (pure SALU).
  const int seg = m0 >> 9;                 // 0:q 1:k 2:v, uniform
  __bf16* __restrict__ dst = (seg == 0) ? qT : ((seg == 1) ? kT : vT);
  const float mult = (seg == 0) ? 0.044194173824159216f : 1.f;  // 512^-0.5
  const int orow  = (m0 & 511) + wm + 8 * g;  // segment-local row base
  const int l_base = n0 + wn + ln;
#pragma unroll
  for (int fi = 0; fi < 2; ++fi) {
#pragma unroll
    for (int r = 0; r < 8; ++r) {
      int oo = orow + 16 * fi + r;
      int h = oo >> 6, d = oo & 63;
      size_t rowbase = ((size_t)b * H_ + h) * L_ * HD_ + d;
#pragma unroll
      for (int fj = 0; fj < 2; ++fj) {
        int l = l_base + 16 * fj;
        dst[rowbase + (size_t)l * HD_] = (__bf16)(acc[fi][fj][r] * mult);
      }
    }
  }
}

// ---------------------------------------------------------------------------
// Stage 2: flash attention per (b,h). WG = 128 query rows (8 waves x 16),
// streaming 16 key/value tiles of 128 columns with online softmax.
// Q and K tiles arrive via the Tensor Data Mover (overlapping the vector-path
// transposed V fill); Out^T[i,d] = sum_j P[i,j] * V^T[j,d].
// ---------------------------------------------------------------------------
__global__ __launch_bounds__(256) void attn_kernel(
    const __bf16* __restrict__ qT, const __bf16* __restrict__ kT,
    const __bf16* __restrict__ vT, float* __restrict__ out)
{
  __shared__ __bf16 sQ[128][72];       // [i][d]  (TDM-padded rows)
  __shared__ __bf16 sK[128][72];       // [j][d]  (TDM-padded rows)
  __shared__ __bf16 sV[64][136];       // transposed: [d][j]
  __shared__ __bf16 sP[8][16][136];    // per-wave P staging [16 rows][128 cols]

  const int bh = blockIdx.y;
  const int i0 = blockIdx.x * 128;
  const int tid  = threadIdx.x;
  const int lane = tid & 31, wv = tid >> 5;
  const int g = lane >> 4, ln = lane & 15;
  const size_t base = (size_t)bh * L_ * HD_;

  // Q tile via TDM (one wave issues; TENSORcnt drained before the barrier)
  if (wv == 0) {
    tdm_load_tile_128x64(qT + base + (size_t)i0 * HD_, ldsoff(&sQ[0][0]));
    __builtin_amdgcn_s_wait_tensorcnt(0);
  }
  __syncthreads();

  // Per-wave Q A-fragments (kept live across the whole j loop)
  v16bf aq[2];
#pragma unroll
  for (int kk = 0; kk < 2; ++kk) {
    const __bf16* p = &sQ[wv * 16 + ln][kk * 32 + g * 8];
    aq[kk] = cat8(*(const v8bf*)p, *(const v8bf*)(p + 16));
  }

  float mrow[8], lrow[8];
  v8f oacc[4];
#pragma unroll
  for (int r = 0; r < 8; ++r) { mrow[r] = -1e30f; lrow[r] = 0.f; }
#pragma unroll
  for (int nf = 0; nf < 4; ++nf)
#pragma unroll
    for (int e = 0; e < 8; ++e) oacc[nf][e] = 0.f;

  for (int j0 = 0; j0 < L_; j0 += 128) {
    __syncthreads();   // previous iteration's sK/sV readers are done
    // K tile [j][d] via TDM (overlaps the vector-path V fill below)
    if (wv == 0)
      tdm_load_tile_128x64(kT + base + (size_t)j0 * HD_, ldsoff(&sK[0][0]));
    // Prefetch next iteration's K/V tiles into cache
    if (j0 + 128 < L_) {
      __builtin_prefetch(kT + base + (size_t)(j0 + 128) * HD_, 0, 1);
      __builtin_prefetch(vT + base + (size_t)(j0 + 128) * HD_, 0, 1);
    }
    // V tile transposed into sV[d][j] so k(=j)-pairs are contiguous
#pragma unroll
    for (int it = 0; it < 4; ++it) {
      int i = tid + it * 256;
      int r = i >> 3, c = i & 7;
      v8bf v = *(const v8bf*)(vT + base + (size_t)(j0 + r) * HD_ + c * 8);
#pragma unroll
      for (int e = 0; e < 8; ++e) sV[c * 8 + e][r] = v[e];
    }
    if (wv == 0) __builtin_amdgcn_s_wait_tensorcnt(0);
    __syncthreads();

    // S = Q K^T : wave computes 16 x 128 in 8 f32 fragments
    v8f s[8];
#pragma unroll
    for (int nf = 0; nf < 8; ++nf) {
      v8f c;
#pragma unroll
      for (int e = 0; e < 8; ++e) c[e] = 0.f;
#pragma unroll
      for (int kk = 0; kk < 2; ++kk) {
        const __bf16* pb = &sK[16 * nf + ln][kk * 32 + g * 8];
        v16bf bb = cat8(*(const v8bf*)pb, *(const v8bf*)(pb + 16));
        c = wmma_bf16(aq[kk], bb, c);
      }
      s[nf] = c;
    }

    // Online softmax. C-layout: row m = r+8g shared by lanes {16g..16g+15}.
    float mnew[8], alpha[8];
#pragma unroll
    for (int r = 0; r < 8; ++r) {
      float t = s[0][r];
#pragma unroll
      for (int nf = 1; nf < 8; ++nf) t = fmaxf(t, s[nf][r]);
#pragma unroll
      for (int off = 1; off < 16; off <<= 1) t = fmaxf(t, __shfl_xor(t, off, 32));
      mnew[r]  = fmaxf(mrow[r], t);
      alpha[r] = __expf(mrow[r] - mnew[r]);
      lrow[r] *= alpha[r];
    }
#pragma unroll
    for (int nf = 0; nf < 4; ++nf)
#pragma unroll
      for (int r = 0; r < 8; ++r) oacc[nf][r] *= alpha[r];

    // P = exp(S - mnew); stage to LDS (C-layout -> row-major) for A-fragments
#pragma unroll
    for (int nf = 0; nf < 8; ++nf) {
#pragma unroll
      for (int r = 0; r < 8; ++r) {
        float p = __expf(s[nf][r] - mnew[r]);
        s[nf][r] = p;
        sP[wv][8 * g + r][16 * nf + ln] = (__bf16)p;
      }
    }
#pragma unroll
    for (int r = 0; r < 8; ++r) {
      float t = 0.f;
#pragma unroll
      for (int nf = 0; nf < 8; ++nf) t += s[nf][r];
#pragma unroll
      for (int off = 1; off < 16; off <<= 1) t += __shfl_xor(t, off, 32);
      lrow[r] += t;
      mrow[r]  = mnew[r];
    }

    // O^T += P(16x128) * V^T(128x64): 4 k-steps x 4 n-frags
    // (same-wave LDS store->load: DS ops in-order per wave; per-wave region)
#pragma unroll
    for (int kk = 0; kk < 4; ++kk) {
      const __bf16* pa = &sP[wv][ln][kk * 32 + g * 8];
      v16bf av = cat8(*(const v8bf*)pa, *(const v8bf*)(pa + 16));
#pragma unroll
      for (int nf = 0; nf < 4; ++nf) {
        const __bf16* pb = &sV[16 * nf + ln][kk * 32 + g * 8];
        v16bf bv = cat8(*(const v8bf*)pb, *(const v8bf*)(pb + 16));
        oacc[nf] = wmma_bf16(av, bv, oacc[nf]);
      }
    }
  }

  // out flat index: ((b*512 + h*64 + d)*2048 + i) == (bh*64 + d)*2048 + i
  const size_t ob = (size_t)bh * HD_ * L_;
#pragma unroll
  for (int nf = 0; nf < 4; ++nf) {
    int d = 16 * nf + ln;
#pragma unroll
    for (int r = 0; r < 8; ++r) {
      int i = i0 + wv * 16 + 8 * g + r;
      out[ob + (size_t)d * L_ + i] = oacc[nf][r] / lrow[r];
    }
  }
}

// ---------------------------------------------------------------------------
extern "C" void kernel_launch(void* const* d_in, const int* in_sizes, int n_in,
                              void* d_out, int out_size, void* d_ws, size_t ws_size,
                              hipStream_t stream) {
  (void)in_sizes; (void)n_in; (void)out_size; (void)ws_size;
  const float* x = (const float*)d_in[0];   // [4,512,2048] f32
  const float* w = (const float*)d_in[1];   // [1536,512]  f32
  float* out = (float*)d_out;               // [4,512,2048] f32

  // Workspace: qT/kT/vT each [32][2048][64] bf16 = 8 MB -> 24 MB total.
  __bf16* qT = (__bf16*)d_ws;
  __bf16* kT = qT + (size_t)B_ * H_ * L_ * HD_;
  __bf16* vT = kT + (size_t)B_ * H_ * L_ * HD_;

  dim3 g1(L_ / 64, (3 * D_) / 128, B_);   // (32, 12, 4)
  qkv_gemm_kernel<<<g1, dim3(256), 0, stream>>>(x, w, qT, kT, vT);

  dim3 g2(L_ / 128, B_ * H_, 1);          // (16, 32)
  attn_kernel<<<g2, dim3(256), 0, stream>>>(qT, kT, vT, out);
}
